// Dynamic_MHGCNLayer_89043261980854
// MI455X (gfx1250) — matrix operand order repaired and checked by reading, not compiled, not run on hardware
//
#include <hip/hip_runtime.h>
#include <hip/hip_bf16.h>

typedef __bf16 bf16_t;
typedef __attribute__((ext_vector_type(16))) __bf16 v16bf;
typedef __attribute__((ext_vector_type(8)))  __bf16 v8bf;
typedef __attribute__((ext_vector_type(8)))  float  v8f;
typedef __attribute__((ext_vector_type(4)))  float  v4f;

#define B_   4
#define N_   2048
#define IN_  256
#define OUT_ 256
#define H_   4
#define HD_  64
#define LN_EPS 1e-5f

__device__ __forceinline__ float lrelu(float x) { return x > 0.f ? x : 0.01f * x; }
__device__ __forceinline__ float sigm(float x)  { return 1.f / (1.f + __expf(-x)); }

// ---- WMMA bf16 fragment loaders (16x16x32, contiguous k order used for both A and B, not the ISA table order) ----
// A matrix 16x32 (MxK): lanes 0-15 hold M=lane, K0-15;
// lanes 16-31 hold M=lane-16, K16-31.  `rowp` = row base + k-chunk.
__device__ __forceinline__ v16bf ldA(const bf16_t* rowp, int lane) {
  const int off = (lane & 16) ? 16 : 0;
  v8bf lo = *(const v8bf*)(rowp + off);
  v8bf hi = *(const v8bf*)(rowp + 8 + off);
  v16bf a;
#pragma unroll
  for (int i = 0; i < 8; ++i) { a[i] = lo[i]; a[8 + i] = hi[i]; }
  return a;
}
// B matrix 32x16 (KxN): lane%16 = column; lanes 0-15 hold K0-15, lanes 16-31 K16-31,
// 2 consecutive K per VGPR.  `colp` = this lane's column base + k-chunk (32 halves).
__device__ __forceinline__ v16bf ldB(const bf16_t* colp, int lane) {
  const bf16_t* p = colp + ((lane & 16) ? 16 : 0);
  v8bf lo = *(const v8bf*)(p);
  v8bf hi = *(const v8bf*)(p + 8);
  v16bf b;
#pragma unroll
  for (int i = 0; i < 8; ++i) { b[i] = lo[i]; b[8 + i] = hi[i]; }
  return b;
}

// ---------------- prep: bf16 convert + weight transpose ----------------
__global__ void k_cvt_feats(const float* __restrict__ nf, bf16_t* __restrict__ xb) {
  int i = blockIdx.x * 256 + threadIdx.x;
  if (i < B_ * N_ * IN_) xb[i] = (bf16_t)nf[i];
}

__global__ void k_wt(const float* __restrict__ Wl, const float* __restrict__ Wq,
                     const float* __restrict__ Wk, bf16_t* __restrict__ wt) {
  int i = blockIdx.x * 256 + threadIdx.x;            // 3 * 256 * 256
  if (i >= 3 * IN_ * OUT_) return;
  int z = i / (IN_ * OUT_), r = i % (IN_ * OUT_);
  int c = r / IN_, kk = r % IN_;                     // wt[z][c][kk] = W[z][kk][c]
  const float* W = (z == 0) ? Wl : (z == 1) ? Wq : Wk;
  wt[i] = (bf16_t)W[kk * OUT_ + c];
}

// ---------------- projections: X (transposed), q, k --------------------
// one wave: 16 rows x 64 cols (one head) of one of the 3 GEMMs
__global__ void __launch_bounds__(32)
k_proj(const bf16_t* __restrict__ xb, const bf16_t* __restrict__ wt,
       const float* __restrict__ b_lin,
       bf16_t* __restrict__ Xt, bf16_t* __restrict__ qw, bf16_t* __restrict__ kw) {
  const int lane = threadIdx.x, ln = lane & 15, hi = (lane >> 4) & 1;
  const int rt = blockIdx.x;            // 512 row tiles over B*N
  const int hcol = blockIdx.y;          // 4 col groups == heads
  const int z = blockIdx.z;             // 0=X, 1=q, 2=k
  const bf16_t* wz = wt + (size_t)z * IN_ * OUT_;
  const int R = rt * 16 + ln;

  v8f acc[4] = {};
  for (int kk0 = 0; kk0 < IN_; kk0 += 32) {
    v16bf a = ldA(xb + (size_t)R * IN_ + kk0, lane);
#pragma unroll
    for (int t = 0; t < 4; ++t) {
      int c = hcol * 64 + t * 16 + ln;
      v16bf b = ldB(wz + (size_t)c * IN_ + kk0, lane);
      acc[t] = __builtin_amdgcn_wmma_f32_16x16x32_bf16(false, a, false, b,
                                                       (short)0, acc[t], false, false);
    }
  }

  const int bidx = (rt * 16) / N_;
  const int n0   = (rt * 16) % N_ + hi * 8;
  if (z == 0) {
    // X stored transposed: [b,h][hd][N]  (contiguous B-frags for A @ X)
#pragma unroll
    for (int t = 0; t < 4; ++t) {
      int d = t * 16 + ln;
      float bias = b_lin[hcol * 64 + d];
      v8bf ov;
#pragma unroll
      for (int e = 0; e < 8; ++e) ov[e] = (bf16_t)lrelu(acc[t][e] + bias);
      *(v8bf*)(Xt + (((size_t)(bidx * H_ + hcol) * HD_ + d) * N_ + n0)) = ov;
    }
  } else {
    bf16_t* dst = (z == 1) ? qw : kw;   // [b,h][N][hd] row-major
#pragma unroll
    for (int t = 0; t < 4; ++t) {
      int d = t * 16 + ln;
#pragma unroll
      for (int e = 0; e < 8; ++e)
        dst[((size_t)(bidx * H_ + hcol) * N_ + n0 + e) * HD_ + d] = (bf16_t)lrelu(acc[t][e]);
    }
  }
}

// ---------------- degree pass: dis = (sum_j adj*sigmoid(qk))^-1/2 ------
__global__ void __launch_bounds__(32)
k_dis(const bf16_t* __restrict__ qw, const bf16_t* __restrict__ kw,
      const float* __restrict__ adj, float* __restrict__ dis) {
  const int lane = threadIdx.x, ln = lane & 15, hi = (lane >> 4) & 1;
  const int it = blockIdx.x, bh = blockIdx.y, b = bh / H_;
  const int i0 = it * 16;
  const bf16_t* qp = qw + (size_t)bh * N_ * HD_;
  const bf16_t* kp = kw + (size_t)bh * N_ * HD_;
  const float*  ap = adj + (size_t)b * N_ * N_;
  const float scale = 0.125f;   // hd^-1/2

  v16bf aq0 = ldA(qp + (size_t)(i0 + ln) * HD_, lane);
  v16bf aq1 = ldA(qp + (size_t)(i0 + ln) * HD_ + 32, lane);

  v8f sum = {};
  for (int j0 = 0; j0 < N_; j0 += 16) {
    v8f s = {};
    v16bf b0 = ldB(kp + (size_t)(j0 + ln) * HD_, lane);
    s = __builtin_amdgcn_wmma_f32_16x16x32_bf16(false, aq0, false, b0, (short)0, s, false, false);
    v16bf b1 = ldB(kp + (size_t)(j0 + ln) * HD_ + 32, lane);
    s = __builtin_amdgcn_wmma_f32_16x16x32_bf16(false, aq1, false, b1, (short)0, s, false, false);
#pragma unroll
    for (int e = 0; e < 8; ++e) {
      int i = i0 + e + 8 * hi;
      sum[e] += ap[(size_t)i * N_ + j0 + ln] * sigm(s[e] * scale);
    }
  }
  // reduce across the 16-lane groups holding a row (C layout: N = lane%16)
#pragma unroll
  for (int m = 1; m < 16; m <<= 1) {
#pragma unroll
    for (int e = 0; e < 8; ++e) sum[e] += __shfl_xor(sum[e], m, 16);
  }
  if (ln == 0) {
    float* dp = dis + (size_t)bh * N_ + i0 + hi * 8;
#pragma unroll
    for (int e = 0; e < 8; ++e) { float dg = sum[e]; dp[e] = dg > 0.f ? rsqrtf(dg) : 0.f; }
  }
}

// ---------------- fused A = D^-1/2 (adj*att) D^-1/2 ;  Y = A @ X -------
__global__ void __launch_bounds__(32)
k_gather(const bf16_t* __restrict__ qw, const bf16_t* __restrict__ kw,
         const bf16_t* __restrict__ Xt, const float* __restrict__ adj,
         const float* __restrict__ dis, float* __restrict__ out) {
  __shared__ __align__(16) bf16_t lds[16 * 32];
  const int lane = threadIdx.x, ln = lane & 15, hi = (lane >> 4) & 1;
  const int it = blockIdx.x, bh = blockIdx.y, b = bh / H_, h = bh % H_;
  const int i0 = it * 16;
  const bf16_t* qp = qw + (size_t)bh * N_ * HD_;
  const bf16_t* kp = kw + (size_t)bh * N_ * HD_;
  const bf16_t* xp = Xt + (size_t)bh * HD_ * N_;
  const float*  ap = adj + (size_t)b * N_ * N_;
  const float*  dp = dis + (size_t)bh * N_;
  const float scale = 0.125f;

  v16bf aq0 = ldA(qp + (size_t)(i0 + ln) * HD_, lane);
  v16bf aq1 = ldA(qp + (size_t)(i0 + ln) * HD_ + 32, lane);

  v8f accY[4] = {};
  for (int j0 = 0; j0 < N_; j0 += 32) {
    // recompute two 16x16 S tiles, normalize, stage 16x32 bf16 tile in LDS
#pragma unroll
    for (int sub = 0; sub < 2; ++sub) {
      const int jb = j0 + sub * 16;
      v8f s = {};
      v16bf b0 = ldB(kp + (size_t)(jb + ln) * HD_, lane);
      s = __builtin_amdgcn_wmma_f32_16x16x32_bf16(false, aq0, false, b0, (short)0, s, false, false);
      v16bf b1 = ldB(kp + (size_t)(jb + ln) * HD_ + 32, lane);
      s = __builtin_amdgcn_wmma_f32_16x16x32_bf16(false, aq1, false, b1, (short)0, s, false, false);
      float dj = dp[jb + ln];
#pragma unroll
      for (int e = 0; e < 8; ++e) {
        int i = i0 + e + 8 * hi;
        float aij = ap[(size_t)i * N_ + jb + ln] * sigm(s[e] * scale) * dj;
        lds[(e + 8 * hi) * 32 + sub * 16 + ln] = (bf16_t)aij;
      }
    }
    asm volatile("s_wait_dscnt 0" ::: "memory");      // LDS RAW fence (same wave)
    // reshape C-layout tile -> A-fragment, multiply by X^T (contiguous B-frags)
    v16bf aA = ldA(lds + ln * 32, lane);
#pragma unroll
    for (int t = 0; t < 4; ++t) {
      int d = t * 16 + ln;
      v16bf bx = ldB(xp + (size_t)d * N_ + j0, lane);
      accY[t] = __builtin_amdgcn_wmma_f32_16x16x32_bf16(false, aA, false, bx,
                                                        (short)0, accY[t], false, false);
    }
  }
  // apply row scale dis_i, scatter to [B,N,256] (pre-LayerNorm)
#pragma unroll
  for (int e = 0; e < 8; ++e) {
    int i = i0 + e + 8 * hi;
    float di = dp[i];
#pragma unroll
    for (int t = 0; t < 4; ++t)
      out[((size_t)(b * N_ + i)) * OUT_ + h * HD_ + t * 16 + ln] = accY[t][e] * di;
  }
}

// ---------------- in-place LayerNorm over 256, one wave per row --------
__global__ void __launch_bounds__(256)
k_ln(float* __restrict__ out, const float* __restrict__ g, const float* __restrict__ be) {
  const int wid = threadIdx.x >> 5, lane = threadIdx.x & 31;
  const int row = blockIdx.x * 8 + wid;
  float* p = out + (size_t)row * OUT_;
  v4f x0 = *(const v4f*)(p + lane * 8);
  v4f x1 = *(const v4f*)(p + lane * 8 + 4);
  float s = 0.f, s2 = 0.f;
#pragma unroll
  for (int e = 0; e < 4; ++e) {
    s  += x0[e] + x1[e];
    s2 += x0[e] * x0[e] + x1[e] * x1[e];
  }
#pragma unroll
  for (int m = 1; m < 32; m <<= 1) { s += __shfl_xor(s, m, 32); s2 += __shfl_xor(s2, m, 32); }
  const float mu  = s * (1.f / 256.f);
  const float var = s2 * (1.f / 256.f) - mu * mu;
  const float r   = rsqrtf(var + LN_EPS);
#pragma unroll
  for (int e = 0; e < 8; ++e) {
    int c = lane * 8 + e;
    float v = (e < 4) ? x0[e] : x1[e - 4];
    p[c] = (v - mu) * r * g[c] + be[c];
  }
}

extern "C" void kernel_launch(void* const* d_in, const int* in_sizes, int n_in,
                              void* d_out, int out_size, void* d_ws, size_t ws_size,
                              hipStream_t stream) {
  const float* nf    = (const float*)d_in[0];
  const float* adj   = (const float*)d_in[1];
  const float* W_lin = (const float*)d_in[2];
  const float* b_lin = (const float*)d_in[3];
  const float* W_q   = (const float*)d_in[4];
  const float* W_k   = (const float*)d_in[5];
  const float* ln_g  = (const float*)d_in[6];
  const float* ln_b  = (const float*)d_in[7];
  float* out = (float*)d_out;

  char* ws = (char*)d_ws;
  size_t off = 0;
  bf16_t* xb = (bf16_t*)(ws + off); off += (size_t)B_ * N_ * IN_ * 2;       // 4 MB
  bf16_t* wt = (bf16_t*)(ws + off); off += (size_t)3 * IN_ * OUT_ * 2;      // 384 KB
  bf16_t* qw = (bf16_t*)(ws + off); off += (size_t)B_ * H_ * N_ * HD_ * 2;  // 4 MB
  bf16_t* kw = (bf16_t*)(ws + off); off += (size_t)B_ * H_ * N_ * HD_ * 2;  // 4 MB
  bf16_t* Xt = (bf16_t*)(ws + off); off += (size_t)B_ * H_ * HD_ * N_ * 2;  // 4 MB
  float*  ds = (float*)(ws + off);  off += (size_t)B_ * H_ * N_ * 4;        // 128 KB

  k_cvt_feats<<<(B_ * N_ * IN_ + 255) / 256, 256, 0, stream>>>(nf, xb);
  k_wt<<<(3 * IN_ * OUT_ + 255) / 256, 256, 0, stream>>>(W_lin, W_q, W_k, wt);
  k_proj<<<dim3((B_ * N_) / 16, H_, 3), 32, 0, stream>>>(xb, wt, b_lin, Xt, qw, kw);
  k_dis<<<dim3(N_ / 16, B_ * H_), 32, 0, stream>>>(qw, kw, adj, ds);
  k_gather<<<dim3(N_ / 16, B_ * H_), 32, 0, stream>>>(qw, kw, Xt, adj, ds, out);
  k_ln<<<(B_ * N_) / 8, 256, 0, stream>>>(out, ln_g, ln_b);
}
